// SDPMOE_50843822850503
// MI455X (gfx1250) — matrix operand order repaired and checked by reading, not compiled, not run on hardware
//
#include <hip/hip_runtime.h>
#include <math.h>

// ---------------------------------------------------------------------------
// MoE (top-2, E=8, D=1024, H=256) for gfx1250 using bf16 WMMA.
// Pipeline: fp32->bf16 conversion / weight transposition, gating MLP with
// WMMA, per-expert up-projection with WMMA, and one big down-projection GEMM
// (K = E*H + E gate/bias columns + pad = 2080) with WMMA.
// Fragment loads are software-pipelined: the next B fragment (and next-k A
// fragment) is issued before the current WMMA so the wave waits on partial
// loadcnt instead of draining to zero per WMMA.
// ---------------------------------------------------------------------------

typedef __attribute__((ext_vector_type(16))) __bf16 v16bf;
typedef __attribute__((ext_vector_type(8)))  float  v8f;

union Frag16 { v16bf v; uint4 q[2]; };   // 16 bf16 = 32 bytes

__device__ __forceinline__ unsigned short f2bf(float f) {
    union { float f; unsigned int u; } c; c.f = f;
    unsigned int u = c.u;
    unsigned int lsb = (u >> 16) & 1u;
    u += 0x7fffu + lsb;                 // round-to-nearest-even
    return (unsigned short)(u >> 16);
}
__device__ __forceinline__ float bf2f(unsigned short h) {
    union { unsigned int u; float f; } c; c.u = ((unsigned int)h) << 16;
    return c.f;
}
__device__ __forceinline__ float gelu_exact(float x) {
    return 0.5f * x * (1.0f + erff(x * 0.70710678118654752f));
}

__device__ __forceinline__ v8f wmma_bf16(const Frag16& a, const Frag16& b, v8f c) {
    return __builtin_amdgcn_wmma_f32_16x16x32_bf16(
        /*neg_a=*/false, a.v, /*neg_b=*/false, b.v,
        /*c_mod=*/(short)0, c, /*reuse_a=*/false, /*reuse_b=*/false);
}

// Load A fragment (16x32 bf16 tile, row-major source).
// Lane l (r = l&15, hi = l>>4): row r, K = hi*16..hi*16+15.
__device__ __forceinline__ Frag16 loadA(const unsigned short* __restrict__ rowptr,
                                        int kbase, int hi) {
    Frag16 f;
    f.q[0] = *(const uint4*)(rowptr + kbase + hi * 16);
    f.q[1] = *(const uint4*)(rowptr + kbase + hi * 16 + 8);
    return f;
}
// Load B fragment (32x16 bf16 tile) from *transposed* weights W^T[N][K]
// (row = output column n, contiguous over K). Lane l: n = l&15, K = hi*16..+15.
__device__ __forceinline__ Frag16 loadB(const unsigned short* __restrict__ wt_row,
                                        int kbase, int hi) {
    Frag16 f;
    const uint4* p = (const uint4*)(wt_row + kbase + hi * 16);
    f.q[0] = p[0];
    f.q[1] = p[1];
    return f;
}

// ------------------------- conversion kernels ------------------------------

__global__ void cvt_x_bf16(const float* __restrict__ src,
                           unsigned short* __restrict__ dst, int n) {
    int i = blockIdx.x * blockDim.x + threadIdx.x;
    int stride = gridDim.x * blockDim.x;
    for (; i < n; i += stride) dst[i] = f2bf(src[i]);
}

// w1 : [E][D=1024][H=256] -> w1t : [E][H=256][D=1024] (bf16)
__global__ void build_w1t(const float* __restrict__ w1,
                          unsigned short* __restrict__ w1t) {
    int i = blockIdx.x * blockDim.x + threadIdx.x;
    int stride = gridDim.x * blockDim.x;
    const int n = 8 * 256 * 1024;
    for (; i < n; i += stride) {
        int e = i >> 18;             // / (256*1024)
        int rem = i & 0x3FFFF;
        int h = rem >> 10;           // / 1024
        int dd = rem & 1023;
        w1t[i] = f2bf(w1[e * (1024 * 256) + dd * 256 + h]);
    }
}

// w2 : [E][H=256][D=1024], b2 : [E][D] ->
// w2tcat : [D=1024][K=2080] with K layout: [e*256+j] = w2[e][j][n],
// [2048+e] = b2[e][n], [2056..2079] = 0.
__global__ void build_w2tcat(const float* __restrict__ w2,
                             const float* __restrict__ b2,
                             unsigned short* __restrict__ w2t) {
    int i = blockIdx.x * blockDim.x + threadIdx.x;
    int stride = gridDim.x * blockDim.x;
    const int n = 1024 * 2080;
    for (; i < n; i += stride) {
        int d = i / 2080;
        int kk = i - d * 2080;
        float v;
        if (kk < 2048) {
            int e = kk >> 8, j = kk & 255;
            v = w2[e * (256 * 1024) + j * 1024 + d];
        } else if (kk < 2056) {
            v = b2[(kk - 2048) * 1024 + d];
        } else {
            v = 0.0f;
        }
        w2t[i] = f2bf(v);
    }
}

// wg1 : [D=1024][Hg=256] -> wg1t : [Hg=256][D=1024] (bf16)
__global__ void build_wg1t(const float* __restrict__ wg1,
                           unsigned short* __restrict__ wg1t) {
    int i = blockIdx.x * blockDim.x + threadIdx.x;
    int stride = gridDim.x * blockDim.x;
    const int n = 256 * 1024;
    for (; i < n; i += stride) {
        int h = i >> 10;
        int dd = i & 1023;
        wg1t[i] = f2bf(wg1[dd * 256 + h]);
    }
}

// ------------------------------ gating -------------------------------------
// Block: 128 threads = 4 waves, 64 tokens. Each wave: 16 tokens x Hg=256 via
// 16 WMMA n-tiles, K=1024 (32 k-steps). Then exact-GELU, tiny logit GEMM,
// softmax + top-2. Writes gates (fp32) and the 8 gate columns (+24 zero pad)
// of the extended hidden matrix hcat[T][2080].
__global__ __launch_bounds__(128) void gate_kernel(
    const unsigned short* __restrict__ xb,    // [T][1024] bf16
    const unsigned short* __restrict__ wg1t,  // [256][1024] bf16
    const float* __restrict__ bg1,            // [256]
    const float* __restrict__ wg2,            // [256][8]
    const float* __restrict__ bg2,            // [8]
    float* __restrict__ gates,                // [T][8]
    unsigned short* __restrict__ hcat)        // [T][2080] bf16
{
    __shared__ unsigned short hg[64 * 256];   // bf16 hidden, 32 KB
    __shared__ float lg[64 * 8];

    const int lane = threadIdx.x & 31;
    const int wave = threadIdx.x >> 5;
    const int r = lane & 15;
    const int hi = lane >> 4;
    const int rowbase = blockIdx.x * 64 + wave * 16;

    v8f acc[16];
#pragma unroll
    for (int n = 0; n < 16; ++n) acc[n] = (v8f){};

    const unsigned short* arow = xb + (size_t)(rowbase + r) * 1024;
    const unsigned short* brow0 = wg1t + (size_t)r * 1024;  // n-tile 0 row

    // software pipeline: fa holds A(k), fb holds next B fragment to consume
    Frag16 fa = loadA(arow, 0, hi);
    Frag16 fb = loadB(brow0, 0, hi);
    for (int k = 0; k < 32; ++k) {
        Frag16 fa_next = fa;
        if (k + 1 < 32) fa_next = loadA(arow, (k + 1) * 32, hi);
#pragma unroll
        for (int n = 0; n < 16; ++n) {
            Frag16 fb_next = fb;
            if (n < 15)
                fb_next = loadB(wg1t + (size_t)((n + 1) * 16 + r) * 1024, k * 32, hi);
            else if (k + 1 < 32)
                fb_next = loadB(brow0, (k + 1) * 32, hi);
            acc[n] = wmma_bf16(fa, fb, acc[n]);
            fb = fb_next;
        }
        fa = fa_next;
    }

    // GELU epilogue into LDS (bf16)
#pragma unroll
    for (int n = 0; n < 16; ++n) {
        float bias = bg1[n * 16 + r];
#pragma unroll
        for (int v = 0; v < 8; ++v) {
            int m = v + 8 * hi;
            float val = gelu_exact(acc[n][v] + bias);
            hg[(wave * 16 + m) * 256 + n * 16 + r] = f2bf(val);
        }
    }
    __syncthreads();

    // logits: thread handles (token = tid>>1, 4 experts)
    {
        int tt = threadIdx.x >> 1;
        int e0 = (threadIdx.x & 1) * 4;
        for (int e = e0; e < e0 + 4; ++e) {
            float s = bg2[e];
            for (int j = 0; j < 256; ++j)
                s += bf2f(hg[tt * 256 + j]) * wg2[j * 8 + e];
            lg[tt * 8 + e] = s;
        }
    }
    __syncthreads();

    // softmax + top-2 (k == 2), one thread per token
    if (threadIdx.x < 64) {
        int tt = threadIdx.x;
        size_t t = (size_t)blockIdx.x * 64 + tt;
        float l[8], p[8];
        float mx = -1e30f;
#pragma unroll
        for (int e = 0; e < 8; ++e) { l[e] = lg[tt * 8 + e]; mx = fmaxf(mx, l[e]); }
        float sum = 0.0f;
#pragma unroll
        for (int e = 0; e < 8; ++e) { p[e] = expf(l[e] - mx); sum += p[e]; }
        float inv = 1.0f / sum;
#pragma unroll
        for (int e = 0; e < 8; ++e) p[e] *= inv;
        int i1 = 0;
#pragma unroll
        for (int e = 1; e < 8; ++e) if (p[e] > p[i1]) i1 = e;
        int i2 = (i1 == 0) ? 1 : 0;
#pragma unroll
        for (int e = 0; e < 8; ++e) if (e != i1 && p[e] > p[i2]) i2 = e;
#pragma unroll
        for (int e = 0; e < 8; ++e) {
            float g = (e == i1 || e == i2) ? p[e] : 0.0f;
            gates[t * 8 + e] = g;
            hcat[t * 2080 + 2048 + e] = f2bf(g);   // gate column (bias trick)
        }
#pragma unroll
        for (int j = 0; j < 24; ++j) hcat[t * 2080 + 2056 + j] = 0;  // K pad
    }
}

// --------------------------- expert up-projection --------------------------
// grid = (T/64, E). Block 256 threads = 8 waves. Wave w: mTile = w&3
// (16 tokens), nHalf = w>>2 (128 of 256 hidden cols -> 8 WMMA n-tiles).
// h = GELU(x @ W1_e + b1_e) * gate[:,e] stored bf16 into hcat[:, e*256 ...].
__global__ __launch_bounds__(256) void expert_up_kernel(
    const unsigned short* __restrict__ xb,    // [T][1024]
    const unsigned short* __restrict__ w1t,   // [E][256][1024]
    const float* __restrict__ b1,             // [E][256]
    const float* __restrict__ gates,          // [T][8]
    unsigned short* __restrict__ hcat)        // [T][2080]
{
    const int e = blockIdx.y;
    const int lane = threadIdx.x & 31;
    const int wave = threadIdx.x >> 5;
    const int r = lane & 15;
    const int hi = lane >> 4;
    const int mTile = wave & 3;
    const int nHalf = wave >> 2;
    const int rowbase = blockIdx.x * 64 + mTile * 16;

    const unsigned short* arow = xb + (size_t)(rowbase + r) * 1024;
    const unsigned short* wbase = w1t + (size_t)e * 256 * 1024;
    const unsigned short* brow0 = wbase + (size_t)(nHalf * 128 + r) * 1024;

    v8f acc[8];
#pragma unroll
    for (int n = 0; n < 8; ++n) acc[n] = (v8f){};

    Frag16 fa = loadA(arow, 0, hi);
    Frag16 fb = loadB(brow0, 0, hi);
    for (int k = 0; k < 32; ++k) {
        Frag16 fa_next = fa;
        if (k + 1 < 32) fa_next = loadA(arow, (k + 1) * 32, hi);
#pragma unroll
        for (int n = 0; n < 8; ++n) {
            Frag16 fb_next = fb;
            if (n < 7)
                fb_next = loadB(wbase + (size_t)(nHalf * 128 + (n + 1) * 16 + r) * 1024,
                                k * 32, hi);
            else if (k + 1 < 32)
                fb_next = loadB(brow0, (k + 1) * 32, hi);
            acc[n] = wmma_bf16(fa, fb, acc[n]);
            fb = fb_next;
        }
        fa = fa_next;
    }

    float bias[8];
#pragma unroll
    for (int n = 0; n < 8; ++n) bias[n] = b1[e * 256 + nHalf * 128 + n * 16 + r];

#pragma unroll
    for (int v = 0; v < 8; ++v) {
        size_t t = (size_t)rowbase + v + 8 * hi;
        float g = gates[t * 8 + e];
#pragma unroll
        for (int n = 0; n < 8; ++n) {
            int h = nHalf * 128 + n * 16 + r;
            float val = gelu_exact(acc[n][v] + bias[n]) * g;
            hcat[t * 2080 + e * 256 + h] = f2bf(val);
        }
    }
}

// ------------------------------ down-projection ----------------------------
// y[T][1024] = Hcat[T][2080] @ W2catT^T  (gated b2 bias folded into the GEMM
// through the gate columns). grid = (T/64, D/128). Block = 8 waves; wave w:
// mTile = w&3, nHalf = w>>2 (64 output cols -> 4 WMMA n-tiles). K = 65 steps.
__global__ __launch_bounds__(256) void down_proj_kernel(
    const unsigned short* __restrict__ hcat,   // [T][2080]
    const unsigned short* __restrict__ w2tcat, // [1024][2080]
    float* __restrict__ y)                     // [T][1024]
{
    const int lane = threadIdx.x & 31;
    const int wave = threadIdx.x >> 5;
    const int r = lane & 15;
    const int hi = lane >> 4;
    const int mTile = wave & 3;
    const int nHalf = wave >> 2;
    const int rowbase = blockIdx.x * 64 + mTile * 16;
    const int colbase = blockIdx.y * 128 + nHalf * 64;

    const unsigned short* arow = hcat + (size_t)(rowbase + r) * 2080;
    const unsigned short* brow0 = w2tcat + (size_t)(colbase + r) * 2080;

    v8f acc[4];
#pragma unroll
    for (int n = 0; n < 4; ++n) acc[n] = (v8f){};

    Frag16 fa = loadA(arow, 0, hi);
    Frag16 fb = loadB(brow0, 0, hi);
    for (int k = 0; k < 65; ++k) {
        Frag16 fa_next = fa;
        if (k + 1 < 65) fa_next = loadA(arow, (k + 1) * 32, hi);
#pragma unroll
        for (int n = 0; n < 4; ++n) {
            Frag16 fb_next = fb;
            if (n < 3)
                fb_next = loadB(w2tcat + (size_t)(colbase + (n + 1) * 16 + r) * 2080,
                                k * 32, hi);
            else if (k + 1 < 65)
                fb_next = loadB(brow0, (k + 1) * 32, hi);
            acc[n] = wmma_bf16(fa, fb, acc[n]);
            fb = fb_next;
        }
        fa = fa_next;
    }

#pragma unroll
    for (int n = 0; n < 4; ++n) {
        int d = colbase + n * 16 + r;
#pragma unroll
        for (int v = 0; v < 8; ++v) {
            size_t t = (size_t)rowbase + v + 8 * hi;
            y[t * 1024 + d] = acc[n][v];
        }
    }
}

// ------------------------------- launcher ----------------------------------

extern "C" void kernel_launch(void* const* d_in, const int* in_sizes, int n_in,
                              void* d_out, int out_size, void* d_ws, size_t ws_size,
                              hipStream_t stream) {
    (void)n_in; (void)out_size; (void)ws_size;

    const float* x   = (const float*)d_in[0];  // [T][1024]
    const float* wg1 = (const float*)d_in[1];  // [1024][256]
    const float* bg1 = (const float*)d_in[2];  // [256]
    const float* wg2 = (const float*)d_in[3];  // [256][8]
    const float* bg2 = (const float*)d_in[4];  // [8]
    const float* w1  = (const float*)d_in[5];  // [8][1024][256]
    const float* b1  = (const float*)d_in[6];  // [8][256]
    const float* w2  = (const float*)d_in[7];  // [8][256][1024]
    const float* b2  = (const float*)d_in[8];  // [8][1024]
    float* y = (float*)d_out;

    const int T = in_sizes[0] / 1024;          // 16384 tokens

    // workspace carve-up (256-byte aligned regions)
    char* ws = (char*)d_ws;
    size_t off = 0;
    auto alloc = [&](size_t bytes) {
        char* p = ws + off;
        off = (off + bytes + 255) & ~(size_t)255;
        return p;
    };
    float*          gates  = (float*)         alloc((size_t)T * 8 * 4);
    unsigned short* xb     = (unsigned short*)alloc((size_t)T * 1024 * 2);
    unsigned short* w1t    = (unsigned short*)alloc((size_t)8 * 256 * 1024 * 2);
    unsigned short* w2tcat = (unsigned short*)alloc((size_t)1024 * 2080 * 2);
    unsigned short* wg1t   = (unsigned short*)alloc((size_t)256 * 1024 * 2);
    unsigned short* hcat   = (unsigned short*)alloc((size_t)T * 2080 * 2);

    // 1) conversions / weight transposition
    cvt_x_bf16  <<<4096, 256, 0, stream>>>(x, xb, T * 1024);
    build_w1t   <<<2048, 256, 0, stream>>>(w1, w1t);
    build_w2tcat<<<2048, 256, 0, stream>>>(w2, b2, w2tcat);
    build_wg1t  <<<512,  256, 0, stream>>>(wg1, wg1t);

    // 2) gating (gates + gate columns of hcat)
    gate_kernel<<<T / 64, 128, 0, stream>>>(xb, wg1t, bg1, wg2, bg2, gates, hcat);

    // 3) per-expert up-projection into hcat
    expert_up_kernel<<<dim3(T / 64, 8), 256, 0, stream>>>(xb, w1t, b1, gates, hcat);

    // 4) fused down-projection + gated bias (one GEMM, K = 2080)
    down_proj_kernel<<<dim3(T / 64, 8), 256, 0, stream>>>(hcat, w2tcat, y);
}
